// Encoder_Decoder_1580547975155
// MI455X (gfx1250) — matrix operand order repaired and checked by reading, not compiled, not run on hardware
//
#include <hip/hip_runtime.h>
#include <math.h>

#define H     1024
#define H4    4096
#define T     2048
#define VOUT  32000
#define STEPS 50
#define EOS_ID 2
#define BOS_ID 1
#define ENC_WGS 32

typedef __attribute__((ext_vector_type(16))) __bf16 v16bf;
typedef __attribute__((ext_vector_type(8)))  __bf16 v8bf;
typedef __attribute__((ext_vector_type(8)))  float  v8f;
typedef __attribute__((ext_vector_type(4)))  int    v4i;

// ---- gfx1250 async global->LDS path (guarded; falls back to plain copy) ----
#if defined(__HIP_DEVICE_COMPILE__) && \
    __has_builtin(__builtin_amdgcn_global_load_async_to_lds_b128) && \
    __has_builtin(__builtin_amdgcn_s_wait_asynccnt)
#define HAVE_ASYNC 1
#define AS1 __attribute__((address_space(1)))
#define AS3 __attribute__((address_space(3)))
#define ASYNC_B128(gsrc, ldst) \
  __builtin_amdgcn_global_load_async_to_lds_b128((AS1 v4i*)(gsrc), (AS3 v4i*)(ldst), 0, 0)
#else
#define HAVE_ASYNC 0
#endif

// ---------------- device helpers ----------------

__device__ __forceinline__ float wave_sum(float s) {
#pragma unroll
  for (int o = 16; o > 0; o >>= 1) s += __shfl_down(s, o, 32);
  return s;
}

__device__ __forceinline__ float sigmoidf(float x) { return 1.0f / (1.0f + expf(-x)); }

__device__ __forceinline__ float dot4(float4 a, float4 b) {
  return a.x * b.x + a.y * b.y + a.z * b.z + a.w * b.w;
}

__device__ __forceinline__ unsigned short f2bf(float f) {  // round-to-nearest-even
  unsigned u = __float_as_uint(f);
  u += 0x7FFFu + ((u >> 16) & 1u);
  return (unsigned short)(u >> 16);
}

// global barrier: bar[0]=count, bar[1]=generation (monotone)
__device__ __forceinline__ void gbar(int* bar, int nwg) {
  __threadfence();
  __syncthreads();
  if (threadIdx.x == 0) {
    int g = __hip_atomic_load(&bar[1], __ATOMIC_ACQUIRE, __HIP_MEMORY_SCOPE_AGENT);
    int old = __hip_atomic_fetch_add(&bar[0], 1, __ATOMIC_ACQ_REL, __HIP_MEMORY_SCOPE_AGENT);
    if (old == nwg - 1) {
      __hip_atomic_store(&bar[0], 0, __ATOMIC_RELAXED, __HIP_MEMORY_SCOPE_AGENT);
      __hip_atomic_fetch_add(&bar[1], 1, __ATOMIC_ACQ_REL, __HIP_MEMORY_SCOPE_AGENT);
    } else {
      while (__hip_atomic_load(&bar[1], __ATOMIC_ACQUIRE, __HIP_MEMORY_SCOPE_AGENT) == g)
        __builtin_amdgcn_s_sleep(2);
    }
  }
  __syncthreads();
}

// ---------------- setup kernels ----------------

__global__ void init_state(int* bar) {
  bar[0] = 0;       // barrier count
  bar[1] = 0;       // barrier generation
  bar[2] = BOS_ID;  // wid
  bar[3] = 0;       // done flag
}

__global__ void f32_to_bf16_k(const float* __restrict__ in, unsigned short* __restrict__ out, int n) {
  int i = blockIdx.x * blockDim.x + threadIdx.x;
  if (i < n) out[i] = f2bf(in[i]);
}

__global__ void gather_emb_bf16(const int* __restrict__ ids, const float* __restrict__ table,
                                unsigned short* __restrict__ out) {
  int t = blockIdx.x;
  const float* src = table + (size_t)ids[t] * H;
  unsigned short* dst = out + (size_t)t * H;
  for (int k = threadIdx.x; k < H; k += blockDim.x) dst[k] = f2bf(src[k]);
}

// ---------------- WMMA GEMM: Xp[4096,2048] = We_ih(bf16) x emb^T(bf16) + be ----------------
// Register-blocked: one wave computes a 32x32 output tile (2x2 WMMA tiles, 4 accumulators),
// reusing each A/B fragment twice per k-step. K=1024 in 32 chunks of 32.

__global__ void gemm_xp_wmma(const unsigned short* __restrict__ Aus,  // [4096,1024] bf16
                             const unsigned short* __restrict__ Bus,  // [2048,1024] bf16 (emb rows)
                             const float* __restrict__ bias,          // be[4096]
                             float* __restrict__ C)                   // [4096,2048]
{
  const __bf16* A = (const __bf16*)Aus;
  const __bf16* B = (const __bf16*)Bus;

  const int NT = T / 32;                                  // 64 macro-tiles along N
  int wave = blockIdx.x * (blockDim.x >> 5) + (threadIdx.x >> 5);
  int lane = threadIdx.x & 31;
  int tm = wave / NT, tn = wave % NT;
  int half = lane >> 4;        // 0: lanes 0-15, 1: lanes 16-31
  int r    = lane & 15;

  const __bf16* A0 = A + (size_t)(tm * 32 + r) * H;       // rows m0..m0+15
  const __bf16* A1 = A0 + (size_t)16 * H;                 // rows m0+16..m0+31
  const __bf16* B0 = B + (size_t)(tn * 32 + r) * H;       // cols n0..n0+15
  const __bf16* B1 = B0 + (size_t)16 * H;                 // cols n0+16..n0+31

  v8f acc00 = {}, acc01 = {}, acc10 = {}, acc11 = {};
#pragma unroll 2
  for (int kk = 0; kk < H / 32; ++kk) {
    int kb = kk * 32;
    int ao = kb + half * 8;      // A frag: K = ao+[0..7] and ao+16+[0..7]
    int bo = kb + half * 8;      // B frag: K = bo+[0..7] and bo+16+[0..7]
    v16bf a0 = __builtin_shufflevector(*(const v8bf*)(A0 + ao), *(const v8bf*)(A0 + ao + 16),
                                       0,1,2,3,4,5,6,7,8,9,10,11,12,13,14,15);
    v16bf a1 = __builtin_shufflevector(*(const v8bf*)(A1 + ao), *(const v8bf*)(A1 + ao + 16),
                                       0,1,2,3,4,5,6,7,8,9,10,11,12,13,14,15);
    v16bf b0 = __builtin_shufflevector(*(const v8bf*)(B0 + bo), *(const v8bf*)(B0 + bo + 16),
                                       0,1,2,3,4,5,6,7,8,9,10,11,12,13,14,15);
    v16bf b1 = __builtin_shufflevector(*(const v8bf*)(B1 + bo), *(const v8bf*)(B1 + bo + 16),
                                       0,1,2,3,4,5,6,7,8,9,10,11,12,13,14,15);

    acc00 = __builtin_amdgcn_wmma_f32_16x16x32_bf16(false, a0, false, b0, (short)0, acc00, false, false);
    acc01 = __builtin_amdgcn_wmma_f32_16x16x32_bf16(false, a0, false, b1, (short)0, acc01, false, false);
    acc10 = __builtin_amdgcn_wmma_f32_16x16x32_bf16(false, a1, false, b0, (short)0, acc10, false, false);
    acc11 = __builtin_amdgcn_wmma_f32_16x16x32_bf16(false, a1, false, b1, (short)0, acc11, false, false);
  }

  // D layout: VGPR v -> row base + half*8 + v, col base + r
  int n0 = tn * 32 + r, n1 = n0 + 16;
#pragma unroll
  for (int v = 0; v < 8; ++v) {
    int m0 = tm * 32 + half * 8 + v;
    int m1 = m0 + 16;
    float bi0 = bias[m0], bi1 = bias[m1];
    C[(size_t)m0 * T + n0] = acc00[v] + bi0;
    C[(size_t)m0 * T + n1] = acc01[v] + bi0;
    C[(size_t)m1 * T + n0] = acc10[v] + bi1;
    C[(size_t)m1 * T + n1] = acc11[v] + bi1;
  }
}

// ---------------- encoder recurrence (persistent, 32 WGs x 256, 1 barrier/step) ----------
// Wave w owns gate indices j = 4w..4w+3; it computes all 4 gate dots for each j, applies
// the LSTM cell, and keeps c[j] in registers for the entire scan.

__global__ void encoder_scan(const float* __restrict__ Whh, const float* __restrict__ Xp,
                             float* __restrict__ hs, float* __restrict__ h, float* __restrict__ c,
                             int* bar)
{
  __shared__ float sh[H];
  const int tid  = threadIdx.x;
  const int gtid = blockIdx.x * 256 + tid;
  const int lane = tid & 31;
  const int wave = gtid >> 5;                 // 0..255 global waves
  const int j0   = wave * 4;                  // 4 owned gate indices

  float creg[4] = {0.0f, 0.0f, 0.0f, 0.0f};   // cell state (valid on lane 0)
  for (int j = gtid; j < H; j += ENC_WGS * 256) h[j] = 0.0f;
  gbar(bar, ENC_WGS);

  for (int t = 0; t < T; ++t) {
    // stage h into LDS (async direct-to-LDS on gfx1250)
#if HAVE_ASYNC
    ASYNC_B128(h + tid * 4, sh + tid * 4);
    __builtin_amdgcn_s_wait_asynccnt(0);
#else
    for (int j = tid; j < H; j += 256) sh[j] = h[j];
#endif
    __syncthreads();
    const float4* sh4 = (const float4*)sh;

#pragma unroll
    for (int jj = 0; jj < 4; ++jj) {
      int j = j0 + jj;
      const float4* w0 = (const float4*)(Whh + (size_t)j * H);
      const float4* w1 = (const float4*)(Whh + (size_t)(j + H) * H);
      const float4* w2 = (const float4*)(Whh + (size_t)(j + 2 * H) * H);
      const float4* w3 = (const float4*)(Whh + (size_t)(j + 3 * H) * H);
      float d0 = 0.0f, d1 = 0.0f, d2 = 0.0f, d3 = 0.0f;
#pragma unroll 2
      for (int k = lane; k < H / 4; k += 32) {
        float4 x = sh4[k];
        d0 += dot4(w0[k], x);
        d1 += dot4(w1[k], x);
        d2 += dot4(w2[k], x);
        d3 += dot4(w3[k], x);
      }
      d0 = wave_sum(d0); d1 = wave_sum(d1); d2 = wave_sum(d2); d3 = wave_sum(d3);
      if (lane == 0) {
        float iv = sigmoidf(d0 + Xp[(size_t)j * T + t]);
        float fv = sigmoidf(d1 + Xp[(size_t)(j + H) * T + t]);
        float gv = tanhf  (d2 + Xp[(size_t)(j + 2 * H) * T + t]);
        float ov = sigmoidf(d3 + Xp[(size_t)(j + 3 * H) * T + t]);
        float cc = fv * creg[jj] + iv * gv;
        creg[jj] = cc;
        float hh = ov * tanhf(cc);
        h[j] = hh;
        hs[(size_t)t * H + j] = hh;
      }
    }
    gbar(bar, ENC_WGS);   // single barrier per step: h[t] visible before step t+1
  }

  if (lane == 0) {
#pragma unroll
    for (int jj = 0; jj < 4; ++jj) c[j0 + jj] = creg[jj];
  }
}

// ---------------- decoder: fused LSTM cell step (1024 waves, wave owns j) ----------------
// Reads hsrc (previous ht), writes hdst (new ht) -- ping-pong avoids cross-block races.

__global__ void dec_step(const float* __restrict__ Wih, const float* __restrict__ Whh,
                         const float* __restrict__ bd, const float* __restrict__ embT,
                         const int* __restrict__ wid,
                         const float* __restrict__ hsrc, float* __restrict__ hdst,
                         float* __restrict__ c)
{
  int wave = (blockIdx.x * blockDim.x + threadIdx.x) >> 5;   // 0..1023 == j
  int lane = threadIdx.x & 31;
  if (wave >= H) return;
  int j = wave;
  const float4* x4 = (const float4*)(embT + (size_t)(*wid) * H);
  const float4* h4 = (const float4*)hsrc;
  const float4* wi0 = (const float4*)(Wih + (size_t)j * H);
  const float4* wi1 = (const float4*)(Wih + (size_t)(j + H) * H);
  const float4* wi2 = (const float4*)(Wih + (size_t)(j + 2 * H) * H);
  const float4* wi3 = (const float4*)(Wih + (size_t)(j + 3 * H) * H);
  const float4* wh0 = (const float4*)(Whh + (size_t)j * H);
  const float4* wh1 = (const float4*)(Whh + (size_t)(j + H) * H);
  const float4* wh2 = (const float4*)(Whh + (size_t)(j + 2 * H) * H);
  const float4* wh3 = (const float4*)(Whh + (size_t)(j + 3 * H) * H);

  float d0 = 0.0f, d1 = 0.0f, d2 = 0.0f, d3 = 0.0f;
#pragma unroll 2
  for (int k = lane; k < H / 4; k += 32) {
    float4 x = x4[k], hh = h4[k];
    d0 += dot4(wi0[k], x) + dot4(wh0[k], hh);
    d1 += dot4(wi1[k], x) + dot4(wh1[k], hh);
    d2 += dot4(wi2[k], x) + dot4(wh2[k], hh);
    d3 += dot4(wi3[k], x) + dot4(wh3[k], hh);
  }
  d0 = wave_sum(d0); d1 = wave_sum(d1); d2 = wave_sum(d2); d3 = wave_sum(d3);
  if (lane == 0) {
    float iv = sigmoidf(d0 + bd[j]);
    float fv = sigmoidf(d1 + bd[j + H]);
    float gv = tanhf  (d2 + bd[j + 2 * H]);
    float ov = sigmoidf(d3 + bd[j + 3 * H]);
    float cc = fv * c[j] + iv * gv;
    c[j] = cc;
    hdst[j] = ov * tanhf(cc);
  }
}

__global__ void attn_scores(const float* __restrict__ hs, const float* __restrict__ h,
                            float* __restrict__ sc)
{
  int wave = (blockIdx.x * blockDim.x + threadIdx.x) >> 5;   // 0..2047
  int lane = threadIdx.x & 31;
  if (wave >= T) return;
  const float4* r4 = (const float4*)(hs + (size_t)wave * H);
  const float4* h4 = (const float4*)h;
  float s = 0.0f;
#pragma unroll 4
  for (int k = lane; k < H / 4; k += 32) s += dot4(r4[k], h4[k]);
  s = wave_sum(s);
  if (lane == 0) sc[wave] = s;
}

__global__ void softmax_k(const float* __restrict__ sc, float* __restrict__ aw) {
  __shared__ float red[1024];
  int tid = threadIdx.x;
  float m = -3.4e38f;
  for (int t = tid; t < T; t += 1024) m = fmaxf(m, sc[t]);
  red[tid] = m; __syncthreads();
  for (int o = 512; o > 0; o >>= 1) { if (tid < o) red[tid] = fmaxf(red[tid], red[tid + o]); __syncthreads(); }
  float mx = red[0]; __syncthreads();
  float z = 0.0f;
  for (int t = tid; t < T; t += 1024) z += expf(sc[t] - mx);
  red[tid] = z; __syncthreads();
  for (int o = 512; o > 0; o >>= 1) { if (tid < o) red[tid] += red[tid + o]; __syncthreads(); }
  float Z = red[0]; __syncthreads();
  for (int t = tid; t < T; t += 1024) aw[t] = expf(sc[t] - mx) / Z;
}

__global__ void context_k(const float* __restrict__ aw, const float* __restrict__ hs,
                          float* __restrict__ ctx)
{
  __shared__ float sa[T];
  int k = threadIdx.x;  // 1024 threads, one output each
#if HAVE_ASYNC
  if (k < 512) ASYNC_B128(aw + k * 4, sa + k * 4);
  __builtin_amdgcn_s_wait_asynccnt(0);
#else
  for (int t = k; t < T; t += 1024) sa[t] = aw[t];
#endif
  __syncthreads();
  float acc = 0.0f;
#pragma unroll 4
  for (int t = 0; t < T; ++t) acc += sa[t] * hs[(size_t)t * H + k];
  ctx[k] = acc;
}

__global__ void attn_combine(const float* __restrict__ Wa, const float* __restrict__ ba,
                             const float* __restrict__ ctx, const float* __restrict__ h,
                             float* __restrict__ htnew)
{
  int wave = (blockIdx.x * blockDim.x + threadIdx.x) >> 5;   // 0..1023
  int lane = threadIdx.x & 31;
  if (wave >= H) return;
  const float* wrow = Wa + (size_t)wave * 2 * H;
  float s = 0.0f;
  for (int k = lane; k < 2 * H; k += 32) {
    float cc = (k < H) ? ctx[k] : h[k - H];
    s += wrow[k] * cc;
  }
  s = wave_sum(s);
  if (lane == 0) htnew[wave] = tanhf(s + ba[wave]);
}

__global__ void logits_k(const float* __restrict__ Wo, const float* __restrict__ bo,
                         const float* __restrict__ ht, const int* __restrict__ done,
                         float* __restrict__ outrow)
{
  int wave = (blockIdx.x * blockDim.x + threadIdx.x) >> 5;   // 0..31999
  int lane = threadIdx.x & 31;
  if (wave >= VOUT) return;
  const float4* wr4 = (const float4*)(Wo + (size_t)wave * H);
  const float4* h4  = (const float4*)ht;
  float s = 0.0f;
#pragma unroll 4
  for (int k = lane; k < H / 4; k += 32) {
    __builtin_prefetch(&wr4[k + 64], 0, 1);   // global_prefetch_b8
    s += dot4(wr4[k], h4[k]);
  }
  s = wave_sum(s);
  if (lane == 0) {
    float l = s + bo[wave];
    outrow[wave] = (*done) ? 0.0f : l;
  }
}

__global__ void argmax_part(const float* __restrict__ l, float* __restrict__ pv, int* __restrict__ pi) {
  __shared__ float sv[256];
  __shared__ int   si[256];
  int tid = threadIdx.x;
  int i = blockIdx.x * 256 + tid;     // 125*256 == 32000 exactly
  sv[tid] = l[i]; si[tid] = i;
  __syncthreads();
  for (int o = 128; o > 0; o >>= 1) {
    if (tid < o) {
      if (sv[tid + o] > sv[tid] || (sv[tid + o] == sv[tid] && si[tid + o] < si[tid])) {
        sv[tid] = sv[tid + o]; si[tid] = si[tid + o];
      }
    }
    __syncthreads();
  }
  if (tid == 0) { pv[blockIdx.x] = sv[0]; pi[blockIdx.x] = si[0]; }
}

__global__ void argmax_final(const float* __restrict__ pv, const int* __restrict__ pi, int nparts,
                             int* wid, int* done, float* __restrict__ out_tok)
{
  __shared__ float sv[128];
  __shared__ int   si[128];
  int tid = threadIdx.x;
  float v = -3.4e38f; int ix = 0x7fffffff;
  for (int b = tid; b < nparts; b += 128) {
    float bv = pv[b]; int bi = pi[b];
    if (bv > v || (bv == v && bi < ix)) { v = bv; ix = bi; }
  }
  sv[tid] = v; si[tid] = ix;
  __syncthreads();
  for (int o = 64; o > 0; o >>= 1) {
    if (tid < o) {
      if (sv[tid + o] > sv[tid] || (sv[tid + o] == sv[tid] && si[tid + o] < si[tid])) {
        sv[tid] = sv[tid + o]; si[tid] = si[tid + o];
      }
    }
    __syncthreads();
  }
  if (tid == 0) {
    int nid = si[0];
    int dn  = *done;
    out_tok[0] = dn ? 0.0f : (float)nid;
    *wid = nid;
    if (nid == EOS_ID) *done = 1;
  }
}

// ---------------- host launch ----------------

extern "C" void kernel_launch(void* const* d_in, const int* in_sizes, int n_in,
                              void* d_out, int out_size, void* d_ws, size_t ws_size,
                              hipStream_t stream) {
  (void)in_sizes; (void)n_in; (void)out_size; (void)ws_size;
  const int*   src_ids      = (const int*)  d_in[0];
  const float* embed_input  = (const float*)d_in[1];
  const float* We_ih        = (const float*)d_in[2];
  const float* We_hh        = (const float*)d_in[3];
  const float* be           = (const float*)d_in[4];
  const float* embed_target = (const float*)d_in[5];
  const float* Wd_ih        = (const float*)d_in[6];
  const float* Wd_hh        = (const float*)d_in[7];
  const float* bd           = (const float*)d_in[8];
  const float* W_attn       = (const float*)d_in[9];
  const float* b_attn       = (const float*)d_in[10];
  const float* W_out        = (const float*)d_in[11];
  const float* b_out        = (const float*)d_in[12];
  float* out = (float*)d_out;            // [0..49] tokens, [50..] logits 50x32000

  char* p = (char*)d_ws;
  auto alloc = [&](size_t bytes) { char* r = p; p += (bytes + 255) & ~(size_t)255; return r; };
  int*            bar    = (int*)           alloc(16);                    // count, gen, wid, done
  unsigned short* embB   = (unsigned short*)alloc((size_t)T * H * 2);     // emb bf16
  unsigned short* WeihB  = (unsigned short*)alloc((size_t)H4 * H * 2);    // We_ih bf16
  float*          Xp     = (float*)         alloc((size_t)H4 * T * 4);    // input projections
  float*          hs     = (float*)         alloc((size_t)T * H * 4);     // encoder states
  float*          hv_a   = (float*)         alloc(H * 4);                 // h ping
  float*          hv_b   = (float*)         alloc(H * 4);                 // h pong
  float*          cv     = (float*)         alloc(H * 4);
  float*          sc     = (float*)         alloc(T * 4);
  float*          aw     = (float*)         alloc(T * 4);
  float*          ctx    = (float*)         alloc(H * 4);
  float*          htnew  = (float*)         alloc(H * 4);
  float*          pval   = (float*)         alloc(128 * 4);
  int*            pidx   = (int*)           alloc(128 * 4);

  init_state<<<1, 1, 0, stream>>>(bar);
  f32_to_bf16_k<<<(H4 * H + 255) / 256, 256, 0, stream>>>(We_ih, WeihB, H4 * H);
  gather_emb_bf16<<<T, 256, 0, stream>>>(src_ids, embed_input, embB);

  // 128x64 = 8192 macro-tiles (32x32), 8 waves per 256-thread block
  gemm_xp_wmma<<<(H4 / 32) * (T / 32) / 8, 256, 0, stream>>>(WeihB, embB, be, Xp);

  encoder_scan<<<ENC_WGS, 256, 0, stream>>>(We_hh, Xp, hs, hv_a, cv, bar);

  float* hsrc = hv_a;   // encoder final hidden state
  float* hdst = hv_b;
  for (int step = 0; step < STEPS; ++step) {
    float* outrow = out + STEPS + (size_t)step * VOUT;
    dec_step<<<128, 256, 0, stream>>>(Wd_ih, Wd_hh, bd, embed_target, bar + 2, hsrc, hdst, cv);
    attn_scores<<<256, 256, 0, stream>>>(hs, hdst, sc);
    softmax_k<<<1, 1024, 0, stream>>>(sc, aw);
    context_k<<<1, 1024, 0, stream>>>(aw, hs, ctx);
    attn_combine<<<128, 256, 0, stream>>>(W_attn, b_attn, ctx, hdst, htnew);
    logits_k<<<4000, 256, 0, stream>>>(W_out, b_out, htnew, bar + 3, outrow);
    argmax_part<<<125, 256, 0, stream>>>(outrow, pval, pidx);
    argmax_final<<<1, 128, 0, stream>>>(pval, pidx, 125, bar + 2, bar + 3, out + step);
    float* tmp = hsrc; hsrc = hdst; hdst = tmp;   // ping-pong decoder hidden state
  }
}
